// _GATLayer_31628139167950
// MI455X (gfx1250) — matrix operand order repaired and checked by reading, not compiled, not run on hardware
//
#include <hip/hip_runtime.h>

// Problem constants (match reference)
#define BB 4
#define NN 2048
#define FF 128
#define NH 4
#define DH 32
#define NEG_SLOPE 0.2f

typedef __attribute__((ext_vector_type(16))) __bf16 v16bf;
typedef __attribute__((ext_vector_type(8)))  float  v8f;

union B16x16 { uint4 u[2]; v16bf v; };

// ---------------------------------------------------------------------------
// Kernel 1: h = x @ W (fp32), write hT (bf16, [B,128,N]) + e_src/e_dst fp32.
// One block = 16 rows of one batch. 128 threads, thread f owns output col f.
// ---------------------------------------------------------------------------
__global__ __launch_bounds__(128)
void gat_project(const float* __restrict__ x, const float* __restrict__ W,
                 const float* __restrict__ a,
                 __bf16* __restrict__ hT,
                 float* __restrict__ e_src, float* __restrict__ e_dst)
{
    const int b  = blockIdx.x >> 7;          // 4 batches
    const int n0 = (blockIdx.x & 127) * 16;  // 128 row-tiles
    const int f  = threadIdx.x;              // 0..127

    __shared__ float xs[16][FF];             // x tile, then reused for h tile

    #pragma unroll
    for (int i = 0; i < 16; ++i)
        xs[i][f] = x[((size_t)(b * NN + n0 + i)) * FF + f];
    __syncthreads();

    float acc[16];
    #pragma unroll
    for (int i = 0; i < 16; ++i) acc[i] = 0.f;

    for (int k = 0; k < FF; k += 4) {
        float w0 = W[(k + 0) * FF + f];
        float w1 = W[(k + 1) * FF + f];
        float w2 = W[(k + 2) * FF + f];
        float w3 = W[(k + 3) * FF + f];
        #pragma unroll
        for (int i = 0; i < 16; ++i) {
            float4 xv = *(const float4*)&xs[i][k];   // LDS b128, broadcast
            acc[i] = fmaf(xv.x, w0, acc[i]);
            acc[i] = fmaf(xv.y, w1, acc[i]);
            acc[i] = fmaf(xv.z, w2, acc[i]);
            acc[i] = fmaf(xv.w, w3, acc[i]);
        }
    }

    // hT[b][f][n0..n0+15] : 16 contiguous bf16 (32 B) per thread
    B16x16 hb;
    #pragma unroll
    for (int i = 0; i < 16; ++i) hb.v[i] = (__bf16)acc[i];
    __bf16* dst = hT + ((size_t)(b * FF + f)) * NN + n0;
    *(uint4*)(dst)     = hb.u[0];
    *(uint4*)(dst + 8) = hb.u[1];

    // stash h tile in LDS for the logit reductions
    __syncthreads();
    #pragma unroll
    for (int i = 0; i < 16; ++i) xs[i][f] = acc[i];
    __syncthreads();

    // 128 tasks: row i (16) x head (4) x {src,dst} (2)
    const int i     = f >> 3;
    const int q     = f & 7;
    const int head  = q >> 1;
    const int isDst = q & 1;
    const float* av = a + head * (2 * DH) + isDst * DH;
    float s = 0.f;
    #pragma unroll
    for (int d = 0; d < DH; ++d)
        s = fmaf(xs[i][head * DH + d], av[d], s);
    float* eo = (isDst ? e_dst : e_src);
    eo[((size_t)(b * NH + head)) * NN + n0 + i] = s;
}

// ---------------------------------------------------------------------------
// Kernel 2: fused masked softmax + aggregation, one wave per (b, head, n-tile).
// Pass 1 (branchless): sweep A = row max (fmax/cndmask only), sweep B =
//   sum of exp(e - M) (one v_exp per element; masked -> exp underflows to 0).
// Pass 2: build bf16 alpha directly in WMMA A-layout (all-select, no
//   divergence), v_wmma_f32_16x16x32_bf16 against hT tiles, ELU epilogue.
// ---------------------------------------------------------------------------
__global__ __launch_bounds__(32)
void gat_attn_agg(const int* __restrict__ adj,
                  const __bf16* __restrict__ hT,
                  const float* __restrict__ e_src,
                  const float* __restrict__ e_dst,
                  float* __restrict__ out)
{
    const int bid  = blockIdx.x;
    const int n0   = (bid & 127) * 16;
    const int head = (bid >> 7) & 3;
    const int b    = bid >> 9;

    const int l    = threadIdx.x;   // 0..31
    const int r    = l & 15;        // matrix row this lane serves
    const int half = l >> 4;

    const int*   adjRow = adj   + ((size_t)(b * NN + n0 + r)) * NN;
    const float* edRow  = e_dst + ((size_t)(b * NH + head)) * NN;
    const float  es     = e_src[((size_t)(b * NH + head)) * NN + n0 + r];

    __shared__ float pM[32], pS[32], rowM[16], rowInvS[16];

    // masked leaky logit, fully select-based (no control flow)
    auto logit = [&](int am, float ed) -> float {
        float z = es + ed;
        float e = z > 0.f ? z : NEG_SLOPE * z;   // v_cndmask
        return am ? e : -1e30f;                  // v_cndmask
    };

    const int mBeg = half * (NN / 2);

    // ---- pass 1, sweep A: row max (no exp, no branches) ----
    float M = -1e30f;
    for (int m = mBeg; m < mBeg + NN / 2; m += 4) {
        int4   av = *(const int4*)  (adjRow + m);
        float4 ev = *(const float4*)(edRow + m);
        M = fmaxf(M, logit(av.x, ev.x));
        M = fmaxf(M, logit(av.y, ev.y));
        M = fmaxf(M, logit(av.z, ev.z));
        M = fmaxf(M, logit(av.w, ev.w));
    }

    // ---- pass 1, sweep B: sum of exp(e - M); masked -> underflow to 0 ----
    float S = 0.f;
    for (int m = mBeg; m < mBeg + NN / 2; m += 4) {
        int4   av = *(const int4*)  (adjRow + m);   // L1/L2 hit (just read)
        float4 ev = *(const float4*)(edRow + m);
        S += __expf(logit(av.x, ev.x) - M);
        S += __expf(logit(av.y, ev.y) - M);
        S += __expf(logit(av.z, ev.z) - M);
        S += __expf(logit(av.w, ev.w) - M);
    }

    pM[l] = M; pS[l] = S;
    __syncthreads();
    if (l < 16) {
        float M0 = pM[l], S0 = pS[l], M1 = pM[l + 16], S1 = pS[l + 16];
        float Mm = fmaxf(M0, M1);
        float Ss = S0 * __expf(M0 - Mm) + S1 * __expf(M1 - Mm);
        rowM[l]    = Mm;
        rowInvS[l] = Ss > 0.f ? 1.f / Ss : 0.f;
    }
    __syncthreads();

    // ---- pass 2: WMMA aggregation ----
    const float Mr = rowM[r];
    const float iS = rowInvS[r];
    const int   kbase = half ? 8 : 0;        // A-layout K base for this lane
    const int   dcol  = l & 15;              // B/D column this lane serves

    // B-matrix source: hT[b][head*32 + d][m] ; lane reads K = kbase+{0..7, 16..23}
    const __bf16* pLo = hT + ((size_t)(b * FF + head * DH + dcol)) * NN + half * 8;
    const __bf16* pHi = pLo + (size_t)16 * NN;

    // alpha in bf16: exp executed unconditionally, mask folded into input
    auto prob = [&](int am, float ed) -> __bf16 {
        float p = __expf(logit(am, ed) - Mr) * iS;   // masked: exp(-huge)=0
        return (__bf16)p;
    };

    v8f c_lo = {}; v8f c_hi = {};
    for (int m0 = 0; m0 < NN; m0 += 32) {
        // adj + e_dst for this lane's 16 A-elements: K = kbase+{0..7, 16..23}
        int4   a0 = *(const int4*)  (adjRow + m0 + kbase);
        int4   a1 = *(const int4*)  (adjRow + m0 + kbase + 4);
        int4   a2 = *(const int4*)  (adjRow + m0 + kbase + 16);
        int4   a3 = *(const int4*)  (adjRow + m0 + kbase + 20);
        float4 e0 = *(const float4*)(edRow  + m0 + kbase);
        float4 e1 = *(const float4*)(edRow  + m0 + kbase + 4);
        float4 e2 = *(const float4*)(edRow  + m0 + kbase + 16);
        float4 e3 = *(const float4*)(edRow  + m0 + kbase + 20);

        v16bf A;
        A[0]  = prob(a0.x, e0.x); A[1]  = prob(a0.y, e0.y);
        A[2]  = prob(a0.z, e0.z); A[3]  = prob(a0.w, e0.w);
        A[4]  = prob(a1.x, e1.x); A[5]  = prob(a1.y, e1.y);
        A[6]  = prob(a1.z, e1.z); A[7]  = prob(a1.w, e1.w);
        A[8]  = prob(a2.x, e2.x); A[9]  = prob(a2.y, e2.y);
        A[10] = prob(a2.z, e2.z); A[11] = prob(a2.w, e2.w);
        A[12] = prob(a3.x, e3.x); A[13] = prob(a3.y, e3.y);
        A[14] = prob(a3.z, e3.z); A[15] = prob(a3.w, e3.w);

        B16x16 Blo, Bhi;
        Blo.u[0] = *(const uint4*)(pLo + m0);
        Blo.u[1] = *(const uint4*)(pLo + m0 + 16);
        Bhi.u[0] = *(const uint4*)(pHi + m0);
        Bhi.u[1] = *(const uint4*)(pHi + m0 + 16);

        c_lo = __builtin_amdgcn_wmma_f32_16x16x32_bf16(
                   false, A, false, Blo.v, (short)0, c_lo, false, false);
        c_hi = __builtin_amdgcn_wmma_f32_16x16x32_bf16(
                   false, A, false, Bhi.v, (short)0, c_hi, false, false);
    }

    // ---- epilogue: ELU + store (D layout: row = j + 8*half, col = dcol) ----
    const int nrow = n0 + (half ? 8 : 0);
    #pragma unroll
    for (int j = 0; j < 8; ++j) {
        float v = c_lo[j];
        v = v > 0.f ? v : (__expf(v) - 1.f);
        out[((size_t)(b * NN + nrow + j)) * FF + head * DH + dcol] = v;
        float w = c_hi[j];
        w = w > 0.f ? w : (__expf(w) - 1.f);
        out[((size_t)(b * NN + nrow + j)) * FF + head * DH + 16 + dcol] = w;
    }
}

// ---------------------------------------------------------------------------
extern "C" void kernel_launch(void* const* d_in, const int* in_sizes, int n_in,
                              void* d_out, int out_size, void* d_ws, size_t ws_size,
                              hipStream_t stream)
{
    const float* x   = (const float*)d_in[0];
    const int*   adj = (const int*)  d_in[1];
    const float* W   = (const float*)d_in[2];
    const float* a   = (const float*)d_in[3];
    float*       out = (float*)d_out;

    // Workspace: hT bf16 [B,128,N] (2 MB) | e_src f32 [B,H,N] | e_dst f32 [B,H,N]
    __bf16* hT    = (__bf16*)d_ws;
    float*  e_src = (float*)((char*)d_ws + (size_t)BB * FF * NN * sizeof(__bf16));
    float*  e_dst = e_src + (size_t)BB * NH * NN;

    gat_project<<<dim3(BB * (NN / 16)), dim3(128), 0, stream>>>(
        x, W, a, hT, e_src, e_dst);

    gat_attn_agg<<<dim3(BB * NH * (NN / 16)), dim3(32), 0, stream>>>(
        adj, hT, e_src, e_dst, out);
}
